// Decoder_4638564680172
// MI455X (gfx1250) — compile-verified, matrix operand order checked by reading, not run on hardware
//
#include <hip/hip_runtime.h>

// ---------------- problem constants ----------------
enum : int {
    B_  = 64,      // batch
    S_  = 64,      // seq len
    I_  = 512,     // encoder hidden / input size
    C_  = 512,     // attention dim
    H_  = 1024,    // decoder hidden
    D_  = 512,     // deep-out size (H = 2*D)
    KY_ = 32000,   // vocab
    M_  = 512,     // target embedding dim
};

typedef __attribute__((ext_vector_type(16))) __bf16 v16bf;
typedef __attribute__((ext_vector_type(8)))  float  v8f;

// ---------------------------------------------------------------------------
// Pack a logical KxN f32 matrix into the CDNA5 WMMA bf16 B-operand layout:
// dst index = ((nTile*kTiles + kTile)*32 + lane)*16 + j
//   n = nTile*16 + (lane & 15)
//   k = kTile*32 + ((lane>>4)<<3) + (j<8 ? j : j+8)
// trans==0: B[k][n] = src[k*N + n];  trans==1: B[k][n] = src[n*K + k]
// ---------------------------------------------------------------------------
__global__ void pack_b_kernel(const float* __restrict__ src, __bf16* __restrict__ dst,
                              int K, int N, int trans)
{
    long long id = (long long)blockIdx.x * blockDim.x + threadIdx.x;
    long long total = (long long)K * N;
    if (id >= total) return;
    int j    = (int)(id & 15);
    int lane = (int)((id >> 4) & 31);
    int kTiles = K >> 5;
    long long rest = id >> 9;             // / (16*32)
    int kT = (int)(rest % kTiles);
    int nT = (int)(rest / kTiles);
    int k = kT * 32 + ((lane >> 4) << 3) + ((j < 8) ? j : j + 8);
    int n = nT * 16 + (lane & 15);
    float val = trans ? src[(long long)n * K + k] : src[(long long)k * N + n];
    dst[id] = (__bf16)val;
}

// ---------------------------------------------------------------------------
// Generic GEMM: C(MxN) = A(MxK, f32 row-major) * Bpacked(bf16) [+ bias] [+ C]
// One wave per 16x64 output tile (1 M-tile x 4 N-tiles): the A fragment is
// converted to bf16 once per 32-wide K-chunk and reused by 4 WMMAs fed from
// 4 strided B pointer streams (constant +1024B advance per K-chunk).
// Optional deterministic split-K over gridDim.y: slice y writes its partial
// sums to C + y*cSplitStride (reduced by reduce_splitk afterwards).
// ---------------------------------------------------------------------------
__global__ void gemm_wmma_bf16(const float* __restrict__ A, int lda,
                               const v16bf* __restrict__ Bp,
                               const float* __restrict__ bias,
                               float* __restrict__ C, int ldc,
                               int mTiles, int nGroups, int kTiles,
                               int accumulate, int kSplit, long long cSplitStride)
{
    int wave = (int)((blockIdx.x * blockDim.x + threadIdx.x) >> 5);
    int lane = (int)(threadIdx.x & 31);
    int total = mTiles * nGroups;
    if (wave >= total) return;                 // wave-uniform: EXEC stays all-1s
    int mT = wave % mTiles;
    int nG = wave / mTiles;
    int nT0 = nG * 4;

    int z     = (int)blockIdx.y;
    int ktPer = kTiles / kSplit;
    int kt0   = z * ktPer;
    int ktEnd = kt0 + ktPer;
    C += (long long)z * cSplitStride;

    int m    = mT * 16 + (lane & 15);
    int koff = (lane >> 4) * 8;
    const float* arow = A + (long long)m * lda;

    // Four loop-invariant B streams, each advancing 32 v16bf (1024 B) per chunk.
    const v16bf* bq[4];
#pragma unroll
    for (int q = 0; q < 4; ++q)
        bq[q] = Bp + lane + ((long long)(nT0 + q) * kTiles + kt0) * 32;

    v8f acc[4];
#pragma unroll
    for (int q = 0; q < 4; ++q)
        acc[q] = (v8f){0.f, 0.f, 0.f, 0.f, 0.f, 0.f, 0.f, 0.f};

    for (int kt = kt0; kt < ktEnd; ++kt) {
        const float* ap = arow + kt * 32 + koff;
        if (kt + 1 < ktEnd)
            __builtin_prefetch((const void*)(ap + 32), 0, 1);
        v16bf a;
#pragma unroll
        for (int j = 0; j < 8; ++j) a[j]     = (__bf16)ap[j];
#pragma unroll
        for (int j = 0; j < 8; ++j) a[8 + j] = (__bf16)ap[16 + j];
#pragma unroll
        for (int q = 0; q < 4; ++q) {
            v16bf b = *bq[q];
            bq[q] += 32;
            acc[q] = __builtin_amdgcn_wmma_f32_16x16x32_bf16(
                         false, a, false, b, (short)0, acc[q], false, false);
        }
    }

    int nlane = lane & 15;
    int rbase = mT * 16 + ((lane >> 4) ? 8 : 0);
#pragma unroll
    for (int q = 0; q < 4; ++q) {
        int n    = (nT0 + q) * 16 + nlane;
        float bv = bias ? bias[n] : 0.f;
#pragma unroll
        for (int v = 0; v < 8; ++v) {
            int row = rbase + v;
            float val = acc[q][v] + bv;
            long long idx = (long long)row * ldc + n;
            if (accumulate) val += C[idx];
            C[idx] = val;
        }
    }
}

// Sum split-K partials: outp[i] = sum_z part[z*n + i]
__global__ void reduce_splitk(const float* __restrict__ part, float* __restrict__ outp,
                              int n, int kSplit)
{
    int i = blockIdx.x * blockDim.x + threadIdx.x;
    if (i >= n) return;
    float s = 0.f;
    for (int z = 0; z < kSplit; ++z) s += part[(long long)z * n + i];
    outp[i] = s;
}

// ---------------------------------------------------------------------------
// Bahdanau attention: e[b,s] = tanh(Ws[b,:] + Uh[b,s,:] + bvec) . vvec
// alpha = softmax_s(e);  ctx[b,:] = sum_s alpha[s] * x[b,s,:]
// One block (64 threads) per batch row.
// ---------------------------------------------------------------------------
__global__ void attention_kernel(const float* __restrict__ x,    // (B,S,I)
                                 const float* __restrict__ Uh,   // (B,S,C)
                                 const float* __restrict__ Ws,   // (B,C)
                                 const float* __restrict__ bvec, // (C)
                                 const float* __restrict__ vvec, // (C)
                                 float* __restrict__ ctx)        // (B,I)
{
    __shared__ float sh[S_];
    int b = blockIdx.x;
    int s = threadIdx.x;                 // 0..63
    const float* uh = Uh + ((long long)b * S_ + s) * C_;
    const float* ws = Ws + (long long)b * C_;
    float e = 0.f;
    for (int c = 0; c < C_; ++c)
        e += tanhf(ws[c] + uh[c] + bvec[c]) * vvec[c];
    sh[s] = e;
    __syncthreads();
    float mx = -1e30f;
    for (int i = 0; i < S_; ++i) mx = fmaxf(mx, sh[i]);
    float ex = __expf(e - mx);
    __syncthreads();
    sh[s] = ex;
    __syncthreads();
    float sum = 0.f;
    for (int i = 0; i < S_; ++i) sum += sh[i];
    float alpha = ex / sum;
    __syncthreads();
    sh[s] = alpha;
    __syncthreads();
    for (int i = s; i < I_; i += S_) {
        float acc = 0.f;
        for (int ss = 0; ss < S_; ++ss)
            acc += sh[ss] * x[((long long)b * S_ + ss) * I_ + i];
        ctx[(long long)b * I_ + i] = acc;
    }
}

// ---------------------------------------------------------------------------
// GRU gate combine (gates already include biases):  r,z,n order.
// ---------------------------------------------------------------------------
__global__ void gru_combine(const float* __restrict__ gi, const float* __restrict__ gh,
                            const float* __restrict__ s, float* __restrict__ snew)
{
    int id = blockIdx.x * blockDim.x + threadIdx.x;
    if (id >= B_ * H_) return;
    int b = id / H_, h = id % H_;
    const float* gib = gi + (long long)b * 3 * H_;
    const float* ghb = gh + (long long)b * 3 * H_;
    float r  = 1.f / (1.f + __expf(-(gib[h] + ghb[h])));
    float z  = 1.f / (1.f + __expf(-(gib[H_ + h] + ghb[H_ + h])));
    float nn = tanhf(gib[2 * H_ + h] + r * ghb[2 * H_ + h]);
    snew[id] = (1.f - z) * nn + z * s[id];
}

// ---------------------------------------------------------------------------
// Maxout pool_size=2: tm[b,d] = max(t[b,2d], t[b,2d+1])
// ---------------------------------------------------------------------------
__global__ void maxout_kernel(const float* __restrict__ t, float* __restrict__ tm)
{
    int id = blockIdx.x * blockDim.x + threadIdx.x;
    if (id >= B_ * D_) return;
    int b = id / D_, d = id % D_;
    tm[id] = fmaxf(t[(long long)b * H_ + 2 * d], t[(long long)b * H_ + 2 * d + 1]);
}

// ---------------------------------------------------------------------------
// Row softmax over the 32000-wide vocab; writes both the output slice and the
// y-feedback buffer for the next step. One 256-thread block per batch row.
// ---------------------------------------------------------------------------
__global__ void softmax_vocab(const float* __restrict__ logits,
                              float* __restrict__ out, float* __restrict__ ybuf)
{
    __shared__ float red[256];
    int b = blockIdx.x, tid = threadIdx.x;
    const float* zr = logits + (long long)b * KY_;
    float mx = -1e30f;
    for (int i = tid; i < KY_; i += 256) mx = fmaxf(mx, zr[i]);
    red[tid] = mx; __syncthreads();
    for (int off = 128; off > 0; off >>= 1) {
        if (tid < off) red[tid] = fmaxf(red[tid], red[tid + off]);
        __syncthreads();
    }
    mx = red[0]; __syncthreads();
    float sum = 0.f;
    for (int i = tid; i < KY_; i += 256) sum += __expf(zr[i] - mx);
    red[tid] = sum; __syncthreads();
    for (int off = 128; off > 0; off >>= 1) {
        if (tid < off) red[tid] += red[tid + off];
        __syncthreads();
    }
    float inv = 1.f / red[0];
    for (int i = tid; i < KY_; i += 256) {
        float p = __expf(zr[i] - mx) * inv;
        out[(long long)b * KY_ + i]  = p;
        ybuf[(long long)b * KY_ + i] = p;
    }
}

__global__ void zero_kernel(float* __restrict__ p, int n)
{
    int i = blockIdx.x * blockDim.x + threadIdx.x;
    if (i < n) p[i] = 0.f;
}

// ---------------------------------------------------------------------------
extern "C" void kernel_launch(void* const* d_in, const int* in_sizes, int n_in,
                              void* d_out, int out_size, void* d_ws, size_t ws_size,
                              hipStream_t stream)
{
    (void)in_sizes; (void)n_in; (void)out_size; (void)ws_size;
    const float* x     = (const float*)d_in[0];   // (B,S,I)
    const float* Ey_t  = (const float*)d_in[1];   // (KY,M)
    const float* W     = (const float*)d_in[2];   // (H,C)
    const float* U     = (const float*)d_in[3];   // (I,C)
    const float* bvec  = (const float*)d_in[4];   // (C)
    const float* vvec  = (const float*)d_in[5];   // (C)
    const float* W_ih  = (const float*)d_in[6];   // (3H,C)
    const float* W_hh  = (const float*)d_in[7];   // (3H,H)
    const float* b_ih  = (const float*)d_in[8];   // (3H)
    const float* b_hh  = (const float*)d_in[9];   // (3H)
    const float* U_o   = (const float*)d_in[10];  // (H,H)
    const float* V_o   = (const float*)d_in[11];  // (M,H)
    const float* C_o   = (const float*)d_in[12];  // (I,H)
    const float* W_o   = (const float*)d_in[13];  // (D,KY)
    float* out = (float*)d_out;                   // (S,B,KY)

    // ----- workspace bump allocator (256B aligned) -----
    char* wp = (char*)d_ws;
    auto alloc = [&](size_t bytes) -> void* {
        void* p = wp;
        wp += (bytes + 255) & ~(size_t)255;
        return p;
    };
    const int KSPLIT = 8;
    float* Uh     = (float*)alloc(sizeof(float) * B_ * S_ * C_);
    float* sA     = (float*)alloc(sizeof(float) * B_ * H_);
    float* sB     = (float*)alloc(sizeof(float) * B_ * H_);
    float* ybuf   = (float*)alloc(sizeof(float) * B_ * KY_);
    float* Ws     = (float*)alloc(sizeof(float) * B_ * C_);
    float* ctx    = (float*)alloc(sizeof(float) * B_ * I_);
    float* gi     = (float*)alloc(sizeof(float) * B_ * 3 * H_);
    float* gh     = (float*)alloc(sizeof(float) * B_ * 3 * H_);
    float* ey     = (float*)alloc(sizeof(float) * B_ * M_);
    float* eyPart = (float*)alloc(sizeof(float) * KSPLIT * B_ * M_);
    float* tvec   = (float*)alloc(sizeof(float) * B_ * H_);
    float* tmax   = (float*)alloc(sizeof(float) * B_ * D_);
    float* logits = (float*)alloc(sizeof(float) * B_ * KY_);
    __bf16* pU   = (__bf16*)alloc(sizeof(__bf16) * (size_t)I_ * C_);
    __bf16* pW   = (__bf16*)alloc(sizeof(__bf16) * (size_t)H_ * C_);
    __bf16* pWih = (__bf16*)alloc(sizeof(__bf16) * (size_t)C_ * 3 * H_);
    __bf16* pWhh = (__bf16*)alloc(sizeof(__bf16) * (size_t)H_ * 3 * H_);
    __bf16* pUo  = (__bf16*)alloc(sizeof(__bf16) * (size_t)H_ * H_);
    __bf16* pVo  = (__bf16*)alloc(sizeof(__bf16) * (size_t)M_ * H_);
    __bf16* pCo  = (__bf16*)alloc(sizeof(__bf16) * (size_t)I_ * H_);
    __bf16* pEy  = (__bf16*)alloc(sizeof(__bf16) * (size_t)KY_ * M_);
    __bf16* pWo  = (__bf16*)alloc(sizeof(__bf16) * (size_t)D_ * KY_);

    auto pack = [&](const float* src, __bf16* dst, int K, int N, int trans) {
        long long total = (long long)K * N;
        int blocks = (int)((total + 255) / 256);
        pack_b_kernel<<<blocks, 256, 0, stream>>>(src, dst, K, N, trans);
    };
    // N must be a multiple of 64 (nGroups = N/64)
    auto gemm = [&](const float* A, int lda, const __bf16* Bp, const float* bias,
                    float* Cm, int ldc, int mT, int nGroups, int kT, int accum,
                    int kSplit, long long cStride) {
        int waves  = mT * nGroups;
        int blocks = (waves + 7) / 8;
        dim3 grid(blocks, kSplit);
        gemm_wmma_bf16<<<grid, 256, 0, stream>>>(
            A, lda, (const v16bf*)Bp, bias, Cm, ldc, mT, nGroups, kT,
            accum, kSplit, cStride);
    };

    // ----- one-time: pack weights to WMMA bf16 layout -----
    pack(U,    pU,   I_,  C_,     0);   // B[k][n] = U[k][n]
    pack(W,    pW,   H_,  C_,     0);
    pack(W_ih, pWih, C_,  3 * H_, 1);   // B = W_ih^T
    pack(W_hh, pWhh, H_,  3 * H_, 1);   // B = W_hh^T
    pack(U_o,  pUo,  H_,  H_,     0);
    pack(V_o,  pVo,  M_,  H_,     0);
    pack(C_o,  pCo,  I_,  H_,     0);
    pack(Ey_t, pEy,  KY_, M_,     0);
    pack(W_o,  pWo,  D_,  KY_,    0);

    // ----- init state -----
    zero_kernel<<<(B_ * H_ + 255) / 256, 256, 0, stream>>>(sA, B_ * H_);
    zero_kernel<<<(B_ * KY_ + 255) / 256, 256, 0, stream>>>(ybuf, B_ * KY_);

    // ----- U_h = input_seq @ U : (B*S, I) x (I, C) -----
    gemm(x, I_, pU, nullptr, Uh, C_, (B_ * S_) / 16, C_ / 64, I_ / 32, 0, 1, 0);

    // ----- sequential decode -----
    for (int t = 0; t < S_; ++t) {
        float* scur  = (t & 1) ? sB : sA;
        float* snext = (t & 1) ? sA : sB;

        // W_s = s @ W : (64,1024)x(1024,512)
        gemm(scur, H_, pW, nullptr, Ws, C_, B_ / 16, C_ / 64, H_ / 32, 0, 1, 0);

        // attention -> ctx
        attention_kernel<<<B_, S_, 0, stream>>>(x, Uh, Ws, bvec, vvec, ctx);

        // GRU gates
        gemm(ctx,  I_, pWih, b_ih, gi, 3 * H_, B_ / 16, (3 * H_) / 64, I_ / 32, 0, 1, 0);
        gemm(scur, H_, pWhh, b_hh, gh, 3 * H_, B_ / 16, (3 * H_) / 64, H_ / 32, 0, 1, 0);
        gru_combine<<<(B_ * H_ + 255) / 256, 256, 0, stream>>>(gi, gh, scur, snext);

        // ey = y @ Ey_t : (64,32000)x(32000,512), deterministic split-K
        gemm(ybuf, KY_, pEy, nullptr, eyPart, M_, B_ / 16, M_ / 64, KY_ / 32,
             0, KSPLIT, (long long)B_ * M_);
        reduce_splitk<<<(B_ * M_ + 255) / 256, 256, 0, stream>>>(
            eyPart, ey, B_ * M_, KSPLIT);

        // t = s_new@U_o + ey@V_o + ctx@C_o : (64,1024)
        gemm(snext, H_, pUo, nullptr, tvec, H_, B_ / 16, H_ / 64, H_ / 32, 0, 1, 0);
        gemm(ey,    M_, pVo, nullptr, tvec, H_, B_ / 16, H_ / 64, M_ / 32, 1, 1, 0);
        gemm(ctx,   I_, pCo, nullptr, tvec, H_, B_ / 16, H_ / 64, I_ / 32, 1, 1, 0);

        // maxout -> tmax (64,512)
        maxout_kernel<<<(B_ * D_ + 255) / 256, 256, 0, stream>>>(tvec, tmax);

        // logits = tmax @ W_o : (64,512)x(512,32000)
        gemm(tmax, D_, pWo, nullptr, logits, KY_, B_ / 16, KY_ / 64, D_ / 32, 0, 1, 0);

        // softmax -> out[t], and feedback y for next step
        softmax_vocab<<<B_, 256, 0, stream>>>(
            logits, out + (long long)t * B_ * KY_, ybuf);
    }
}
